// VGTripletLoss_2705829397238
// MI455X (gfx1250) — compile-verified, matrix operand order checked by reading, not run on hardware
//
#include <hip/hip_runtime.h>

#define NN 1024
#define DD 2048
#define PP 8
#define MARGIN 0.3f

#define GN_BYTES   ((size_t)NN * DD * 2)        /* 4 MiB bf16 normalized global */
#define PN_BYTES   ((size_t)NN * PP * DD * 2)   /* 32 MiB bf16 normalized parts */
#define DIST_BYTES ((size_t)NN * NN * 4)        /* 4 MiB f32 dist matrix */

#define LDS_ROW 65                               /* padded row: conflict-free */
#define LDS_PART (32 * LDS_ROW)                  /* 2080 floats per part slab */

typedef __attribute__((ext_vector_type(16))) __bf16 v16bf;
typedef __attribute__((ext_vector_type(8)))  float  v8f;

union Frag16 { uint4 u[2]; v16bf v; };

__device__ __forceinline__ unsigned short f32_to_bf16(float f) {
  unsigned int u = __float_as_uint(f);
  u += 0x7fffu + ((u >> 16) & 1u);   // round-to-nearest-even
  return (unsigned short)(u >> 16);
}

// A-operand layout (16-bit A 16x32, ISA 7.12.2): lane = half*16 + lrow holds row
// M=lrow, K = half*8 + {0..7} in VGPR0-3 and K = 16 + half*8 + {0..7} in VGPR4-7.
// -> two contiguous 16B loads per lane. B loaded identically from the row-major
// "column matrix" (Gram trick: D[m][n] = dot(rowA_m, rowB_n)).
__device__ __forceinline__ v16bf load_frag_bf16(const unsigned short* p) {
  Frag16 f;
  f.u[0] = *reinterpret_cast<const uint4*>(p);
  f.u[1] = *reinterpret_cast<const uint4*>(p + 16);
  return f.v;
}

// ---------------------------------------------------------------------------
// Kernel 1: L2-normalize each 2048-float row, emit bf16.
// Rows [0,1024) = global_feat, rows [1024, 9216) = partial_feat.
// ---------------------------------------------------------------------------
__global__ __launch_bounds__(256) void normalize_rows_kernel(
    const float* __restrict__ gsrc, const float* __restrict__ psrc,
    unsigned short* __restrict__ gdst, unsigned short* __restrict__ pdst) {
  const int row = blockIdx.x;
  const int t   = threadIdx.x;
  const float* src;
  unsigned short* dst;
  if (row < NN) {
    src = gsrc + (size_t)row * DD;
    dst = gdst + (size_t)row * DD;
  } else {
    const int r = row - NN;
    src = psrc + (size_t)r * DD;
    dst = pdst + (size_t)r * DD;
  }
  const float4* s4 = reinterpret_cast<const float4*>(src);
  const float4 v0 = s4[2 * t];
  const float4 v1 = s4[2 * t + 1];
  float s = v0.x * v0.x + v0.y * v0.y + v0.z * v0.z + v0.w * v0.w +
            v1.x * v1.x + v1.y * v1.y + v1.z * v1.z + v1.w * v1.w;

  __shared__ float red[256];
  red[t] = s;
  __syncthreads();
  #pragma unroll
  for (int off = 128; off > 0; off >>= 1) {
    if (t < off) red[t] += red[t + off];
    __syncthreads();
  }
  const float scale = 1.0f / (sqrtf(red[0]) + 1e-12f);

  union { unsigned short h[8]; uint4 u; } ob;
  ob.h[0] = f32_to_bf16(v0.x * scale);
  ob.h[1] = f32_to_bf16(v0.y * scale);
  ob.h[2] = f32_to_bf16(v0.z * scale);
  ob.h[3] = f32_to_bf16(v0.w * scale);
  ob.h[4] = f32_to_bf16(v1.x * scale);
  ob.h[5] = f32_to_bf16(v1.y * scale);
  ob.h[6] = f32_to_bf16(v1.z * scale);
  ob.h[7] = f32_to_bf16(v1.w * scale);
  *reinterpret_cast<uint4*>(dst + 8 * t) = ob.u;
}

// ---------------------------------------------------------------------------
// Kernel 2: one block (9 waves) owns a 32x64 dist tile. Wave w < 8 computes the
// 32x64 Gram sub-matrix for part w, wave 8 for the global features, all via
// v_wmma_f32_16x16x32_bf16 (2 A-frags x 4 B-frags -> 8 WMMA / 12 loads per
// K-step). Raw dots go to padded LDS; then all 288 threads combine the 9 parts
// with overlap weights in a fixed order (deterministic) and write dist.
// ---------------------------------------------------------------------------
__global__ __launch_bounds__(288) void dist_wmma_kernel(
    const float* __restrict__ part_labels,          // [NN][PP]
    const unsigned short* __restrict__ gN,          // [NN][DD] bf16
    const unsigned short* __restrict__ pN,          // [NN*PP][DD] bf16
    float* __restrict__ dist) {                     // [NN][NN]
  __shared__ float lbuf[9 * LDS_PART];              // ~73 KB, fits CDNA5 WGP LDS

  const int t    = threadIdx.x;
  const int lane = t & 31;
  const int w    = t >> 5;       // wave id == part id (8 == global)
  const int hl   = lane >> 4;    // K-half for A/B loads, M-half for C/D
  const int lr   = lane & 15;
  const int i0   = (blockIdx.x >> 4) << 5;   // 32-wide row tile
  const int j0   = (blockIdx.x & 15) << 6;   // 64-wide col tile

  // ---- Gram phase (per wave) ----
  const unsigned short* base;
  size_t ld;
  if (w < 8) { base = pN + (size_t)w * DD; ld = (size_t)PP * DD; }
  else       { base = gN;                  ld = DD; }

  const unsigned short* pa = base + (size_t)(i0 + lr) * ld + hl * 8;
  const unsigned short* pb = base + (size_t)(j0 + lr) * ld + hl * 8;

  const v8f vzero = {};
  v8f acc[2][4];
  #pragma unroll
  for (int a = 0; a < 2; ++a)
    #pragma unroll
    for (int b = 0; b < 4; ++b) acc[a][b] = vzero;

  #pragma unroll 2
  for (int k0 = 0; k0 < DD; k0 += 32) {
    const v16bf a0 = load_frag_bf16(pa + k0);
    const v16bf a1 = load_frag_bf16(pa + 16 * ld + k0);
    const v16bf b0 = load_frag_bf16(pb + k0);
    const v16bf b1 = load_frag_bf16(pb + 16 * ld + k0);
    const v16bf b2 = load_frag_bf16(pb + 32 * ld + k0);
    const v16bf b3 = load_frag_bf16(pb + 48 * ld + k0);
    acc[0][0] = __builtin_amdgcn_wmma_f32_16x16x32_bf16(
        false, a0, false, b0, (short)0, acc[0][0], false, false);
    acc[0][1] = __builtin_amdgcn_wmma_f32_16x16x32_bf16(
        false, a0, false, b1, (short)0, acc[0][1], false, false);
    acc[0][2] = __builtin_amdgcn_wmma_f32_16x16x32_bf16(
        false, a0, false, b2, (short)0, acc[0][2], false, false);
    acc[0][3] = __builtin_amdgcn_wmma_f32_16x16x32_bf16(
        false, a0, false, b3, (short)0, acc[0][3], false, false);
    acc[1][0] = __builtin_amdgcn_wmma_f32_16x16x32_bf16(
        false, a1, false, b0, (short)0, acc[1][0], false, false);
    acc[1][1] = __builtin_amdgcn_wmma_f32_16x16x32_bf16(
        false, a1, false, b1, (short)0, acc[1][1], false, false);
    acc[1][2] = __builtin_amdgcn_wmma_f32_16x16x32_bf16(
        false, a1, false, b2, (short)0, acc[1][2], false, false);
    acc[1][3] = __builtin_amdgcn_wmma_f32_16x16x32_bf16(
        false, a1, false, b3, (short)0, acc[1][3], false, false);
  }

  // C/D layout: VGPR r, lanes 0-15 -> M=r, lanes 16-31 -> M=r+8; N = lane&15.
  float* my = lbuf + w * LDS_PART;
  #pragma unroll
  for (int a = 0; a < 2; ++a)
    #pragma unroll
    for (int b = 0; b < 4; ++b)
      #pragma unroll
      for (int r = 0; r < 8; ++r)
        my[(a * 16 + hl * 8 + r) * LDS_ROW + b * 16 + lr] = acc[a][b][r];

  __syncthreads();

  // ---- Combine phase: fixed part order -> deterministic ----
  for (int e = t; e < 32 * 64; e += 288) {
    const int m = e >> 6;
    const int n = e & 63;
    const float4 laLo = *reinterpret_cast<const float4*>(
        part_labels + (size_t)(i0 + m) * PP);
    const float4 laHi = *reinterpret_cast<const float4*>(
        part_labels + (size_t)(i0 + m) * PP + 4);
    const float4 lbLo = *reinterpret_cast<const float4*>(
        part_labels + (size_t)(j0 + n) * PP);
    const float4 lbHi = *reinterpret_cast<const float4*>(
        part_labels + (size_t)(j0 + n) * PP + 4);
    const float la[8] = {laLo.x, laLo.y, laLo.z, laLo.w,
                         laHi.x, laHi.y, laHi.z, laHi.w};
    const float lb[8] = {lbLo.x, lbLo.y, lbLo.z, lbLo.w,
                         lbHi.x, lbHi.y, lbHi.z, lbHi.w};
    const int le = m * LDS_ROW + n;
    float num = 0.0f;
    float den = 1.0f;
    #pragma unroll
    for (int p = 0; p < 8; ++p) {
      const float ov = la[p] * lb[p];
      num += ov * (1.0f - lbuf[p * LDS_PART + le]) * 0.5f;
      den += ov;
    }
    num += (1.0f - lbuf[8 * LDS_PART + le]) * 0.5f;   // global-feature term
    dist[(size_t)(i0 + m) * NN + (j0 + n)] = num / den;
  }
}

// ---------------------------------------------------------------------------
// Kernel 3: per-row hard positive / hard negative mining + hinge.
// ---------------------------------------------------------------------------
__global__ __launch_bounds__(256) void mine_kernel(
    const float* __restrict__ dist, const int* __restrict__ labels,
    float* __restrict__ rowloss) {
  const int i = blockIdx.x;
  const int t = threadIdx.x;
  const int li = labels[i];
  float ap = -1e30f;
  float an = 1e30f;
  #pragma unroll
  for (int jj = 0; jj < NN / 256; ++jj) {
    const int j = t + jj * 256;
    const float dv = dist[(size_t)i * NN + j];
    if (labels[j] == li) ap = fmaxf(ap, dv);
    else                 an = fminf(an, dv);
  }
  __shared__ float smax[256];
  __shared__ float smin[256];
  smax[t] = ap;
  smin[t] = an;
  __syncthreads();
  #pragma unroll
  for (int off = 128; off > 0; off >>= 1) {
    if (t < off) {
      smax[t] = fmaxf(smax[t], smax[t + off]);
      smin[t] = fminf(smin[t], smin[t + off]);
    }
    __syncthreads();
  }
  if (t == 0) rowloss[i] = fmaxf(0.0f, smax[0] - smin[0] + MARGIN);
}

// ---------------------------------------------------------------------------
// Kernel 4: deterministic final mean (no atomics -> graph-replay safe).
// ---------------------------------------------------------------------------
__global__ __launch_bounds__(256) void final_reduce_kernel(
    const float* __restrict__ rowloss, float* __restrict__ out) {
  const int t = threadIdx.x;
  float s = 0.0f;
  #pragma unroll
  for (int jj = 0; jj < NN / 256; ++jj) s += rowloss[t + jj * 256];
  __shared__ float red[256];
  red[t] = s;
  __syncthreads();
  #pragma unroll
  for (int off = 128; off > 0; off >>= 1) {
    if (t < off) red[t] += red[t + off];
    __syncthreads();
  }
  if (t == 0) out[0] = red[0] * (1.0f / (float)NN);
}

extern "C" void kernel_launch(void* const* d_in, const int* in_sizes, int n_in,
                              void* d_out, int out_size, void* d_ws, size_t ws_size,
                              hipStream_t stream) {
  const float* global_feat   = (const float*)d_in[0];   // [1024][2048] f32
  const float* partial_feat  = (const float*)d_in[1];   // [1024][8][2048] f32
  const float* part_labels   = (const float*)d_in[2];   // [1024][8] f32
  const int*   global_labels = (const int*)d_in[3];     // [1024] int

  char* ws = (char*)d_ws;
  unsigned short* gN  = (unsigned short*)ws;
  unsigned short* pN  = (unsigned short*)(ws + GN_BYTES);
  float* dist    = (float*)(ws + GN_BYTES + PN_BYTES);
  float* rowloss = (float*)(ws + GN_BYTES + PN_BYTES + DIST_BYTES);
  float* out = (float*)d_out;

  // 1) normalize + bf16 convert: 9216 rows, one block each
  normalize_rows_kernel<<<NN + NN * PP, 256, 0, stream>>>(
      global_feat, partial_feat, gN, pN);

  // 2) fused WMMA distance: 32 i-tiles x 16 j-tiles, 9 waves (parts) per block
  dist_wmma_kernel<<<(NN / 32) * (NN / 64), 288, 0, stream>>>(
      part_labels, gN, pN, dist);

  // 3) hard mining per row
  mine_kernel<<<NN, 256, 0, stream>>>(dist, global_labels, rowloss);

  // 4) mean
  final_reduce_kernel<<<1, 256, 0, stream>>>(rowloss, out);
}
